// SatelliteGNN_33792802685612
// MI455X (gfx1250) — matrix operand order repaired and checked by reading, not compiled, not run on hardware
//
#include <hip/hip_runtime.h>
#include <hip/hip_bf16.h>
#include <stdint.h>

// GCLSTM (ChebConv K=1 degenerates to per-node dense recurrence) on gfx1250.
// Strategy: 16-node tiles, persistent H/C on-chip across all 64 timesteps,
// bf16 WMMA (16x16x32) with f32 accumulation for H @ Theta_h.
// Nonlinearities use CDNA5 hardware V_TANH_F32 when available.

typedef __attribute__((ext_vector_type(16))) __bf16 v16bf;
typedef __attribute__((ext_vector_type(8)))  float  v8f;

#define T_STEPS 64
#define NNODES  20000
#define FIN     3
#define HD      64
#define GD      256   // 4*HD, gates fused i|f|c|o
#define TILE_N  16
#define THREADS 128   // 4 waves of 32

#if defined(__has_builtin)
#if __has_builtin(__builtin_amdgcn_tanhf)
#define USE_HW_TANH 1
#endif
#endif

__device__ __forceinline__ uint32_t f2bf_pk(float a, float b) {
    // pack two floats into two bf16 (round-to-nearest-even), low = a
    uint32_t ua = __float_as_uint(a);
    uint32_t ub = __float_as_uint(b);
    ua = (ua + 0x7FFFu + ((ua >> 16) & 1u)) >> 16;
    ub = (ub + 0x7FFFu + ((ub >> 16) & 1u)) >> 16;
    return ua | (ub << 16);
}

__device__ __forceinline__ float ftanh(float x) {
#ifdef USE_HW_TANH
    return __builtin_amdgcn_tanhf(x);          // single V_TANH_F32
#else
    x = fminf(fmaxf(x, -15.f), 15.f);
    float e = __expf(2.f * x);
    return (e - 1.f) / (e + 1.f);
#endif
}
__device__ __forceinline__ float fsig(float x) {
#ifdef USE_HW_TANH
    return __builtin_fmaf(0.5f, __builtin_amdgcn_tanhf(0.5f * x), 0.5f);
#else
    x = fminf(fmaxf(x, -30.f), 30.f);
    return 1.f / (1.f + __expf(-x));
#endif
}

union BFrag { v16bf v; uint32_t u[8]; };

__global__ __launch_bounds__(THREADS) void gclstm_kernel(
    const float* __restrict__ x_seq,   // [T][N][3]
    const float* __restrict__ W_x,     // [3][256]
    const float* __restrict__ Theta,   // [64][256]
    const float* __restrict__ b_conv,  // [256]
    const float* __restrict__ b_gate,  // [256]
    const float* __restrict__ w_ci,    // [64]
    const float* __restrict__ w_cf,    // [64]
    const float* __restrict__ w_co,    // [64]
    float* __restrict__ hsum_out)      // [64], global atomic accumulation
{
    // H tile as bf16 pairs: [16 nodes][64 dims] -> [16][32] uint32
    __shared__ uint32_t sH[TILE_N * HD / 2];
    // gate pre-activations: [4 gates][16 nodes][64 dims] f32
    __shared__ float sG[4 * TILE_N * HD];
    __shared__ float sX[TILE_N * FIN];
    __shared__ float sHsum[HD];

    const int tid   = threadIdx.x;
    const int wave  = tid >> 5;
    const int lane  = tid & 31;
    const int l15   = lane & 15;
    const int lhalf = lane >> 4;
    const int node_base = blockIdx.x * TILE_N;

    // ---------- Preload B fragments: Theta_h columns [wave*64, wave*64+64) as bf16.
    // B (32x16, 16-bit) lane layout: lanes 0-15 hold col N=lane, K=0..15 in pairs
    // across VGPR 0..7; lanes 16-31 hold K=16..31 (contiguous K order chosen here, same as the A fragments).
    BFrag bfrag[4][2];
    const int colbase_w = wave * 64;
    for (int nt = 0; nt < 4; ++nt) {
        const int col = colbase_w + nt * 16 + l15;
        for (int kc = 0; kc < 2; ++kc) {
            const int kb = kc * 32 + lhalf * 16;
            #pragma unroll
            for (int v = 0; v < 8; ++v) {
                const int k = kb + 2 * v;
                bfrag[nt][kc].u[v] = f2bf_pk(Theta[k * GD + col], Theta[(k + 1) * GD + col]);
            }
        }
    }
    // per-column constants (column depends only on lane within this wave)
    float bias[4], wx0[4], wx1[4], wx2[4];
    for (int nt = 0; nt < 4; ++nt) {
        const int col = colbase_w + nt * 16 + l15;
        bias[nt] = b_conv[col] + b_gate[col];
        wx0[nt]  = W_x[0 * GD + col];
        wx1[nt]  = W_x[1 * GD + col];
        wx2[nt]  = W_x[2 * GD + col];
    }

    // ---------- Elementwise-phase assignment: thread owns node (tid>>3),
    // dims [(tid&7)*8, +8). C state lives in registers for the whole sequence.
    const int ew_node = tid >> 3;
    const int ew_dim0 = (tid & 7) * 8;
    float creg[8], hreg[8], wci[8], wcf[8], wco[8];
    #pragma unroll
    for (int j = 0; j < 8; ++j) {
        creg[j] = 0.f; hreg[j] = 0.f;
        wci[j] = w_ci[ew_dim0 + j];
        wcf[j] = w_cf[ew_dim0 + j];
        wco[j] = w_co[ew_dim0 + j];
    }

    for (int i = tid; i < TILE_N * HD / 2; i += THREADS) sH[i] = 0u;  // H0 = 0
    if (tid < HD) sHsum[tid] = 0.f;
    __syncthreads();

    for (int t = 0; t < T_STEPS; ++t) {
        // x tile for this step: 16 nodes x 3 feats
        if (tid < TILE_N * FIN) {
            const int nd = tid / FIN, f = tid % FIN;
            sX[tid] = x_seq[(size_t)t * NNODES * FIN + (size_t)(node_base + nd) * FIN + f];
        }
        __syncthreads();

        // Build A fragments (16x32 bf16) from sH in contiguous K order (same as B):
        // lane l: row M=l&15; VGPR v: K0 = (l>>4)*16 + v*2
        BFrag a0, a1;
        {
            const int M = l15;
            #pragma unroll
            for (int v = 0; v < 8; ++v) {
                const int k0 = (lhalf * 16) + (v * 2);
                a0.u[v] = sH[M * 32 + (k0 >> 1)];
                a1.u[v] = sH[M * 32 + ((32 + k0) >> 1)];
            }
        }
        // per-row x values (row r maps to node r + 8*lhalf in C/D layout)
        float xr[8][3];
        #pragma unroll
        for (int r = 0; r < 8; ++r) {
            const int nl = r + 8 * lhalf;
            xr[r][0] = sX[nl * FIN + 0];
            xr[r][1] = sX[nl * FIN + 1];
            xr[r][2] = sX[nl * FIN + 2];
        }

        // acc init = bias + x_t @ W_x (exact f32), directly in C-matrix layout
        v8f acc[4];
        #pragma unroll
        for (int nt = 0; nt < 4; ++nt)
            #pragma unroll
            for (int r = 0; r < 8; ++r)
                acc[nt][r] = bias[nt] + xr[r][0] * wx0[nt]
                                      + xr[r][1] * wx1[nt]
                                      + xr[r][2] * wx2[nt];

        // H @ Theta: K=64 -> two bf16 WMMAs per 16x16 output tile, 4 tiles/wave
        #pragma unroll
        for (int nt = 0; nt < 4; ++nt)
            acc[nt] = __builtin_amdgcn_wmma_f32_16x16x32_bf16(
                false, a0.v, false, bfrag[nt][0].v, (short)0, acc[nt], false, false);
        #pragma unroll
        for (int nt = 0; nt < 4; ++nt)
            acc[nt] = __builtin_amdgcn_wmma_f32_16x16x32_bf16(
                false, a1.v, false, bfrag[nt][1].v, (short)0, acc[nt], false, false);

        // scatter gate pre-activations: wave w == gate w
        #pragma unroll
        for (int nt = 0; nt < 4; ++nt) {
            const int dim = nt * 16 + l15;
            #pragma unroll
            for (int r = 0; r < 8; ++r) {
                const int nl = r + 8 * lhalf;
                sG[wave * (TILE_N * HD) + nl * HD + dim] = acc[nt][r];
            }
        }
        __syncthreads();

        // elementwise LSTM cell update (C in registers, H back to LDS as bf16)
        #pragma unroll
        for (int j = 0; j < 8; ++j) {
            const int dm = ew_dim0 + j;
            const float gi = sG[0 * 1024 + ew_node * 64 + dm];
            const float gf = sG[1 * 1024 + ew_node * 64 + dm];
            const float gc = sG[2 * 1024 + ew_node * 64 + dm];
            const float go = sG[3 * 1024 + ew_node * 64 + dm];
            const float c0 = creg[j];
            const float I  = fsig(gi + wci[j] * c0);
            const float Fg = fsig(gf + wcf[j] * c0);
            const float Tc = ftanh(gc);
            const float Cn = Fg * c0 + I * Tc;
            const float O  = fsig(go + wco[j] * Cn);
            const float Hn = O * ftanh(Cn);
            creg[j] = Cn;
            hreg[j] = Hn;
        }
        #pragma unroll
        for (int j = 0; j < 8; j += 2)
            sH[ew_node * 32 + ((ew_dim0 + j) >> 1)] = f2bf_pk(hreg[j], hreg[j + 1]);
        __syncthreads();
    }

    // mean-pool partial sum over this block's 16 nodes, then one global atomic/dim
    #pragma unroll
    for (int j = 0; j < 8; ++j)
        atomicAdd(&sHsum[ew_dim0 + j], hreg[j]);
    __syncthreads();
    if (tid < HD) atomicAdd(&hsum_out[tid], sHsum[tid]);
}

__global__ void finalize_kernel(const float* __restrict__ hsum,
                                const float* __restrict__ W_lin,  // [64][2]
                                const float* __restrict__ b_lin,  // [2]
                                float* __restrict__ out)          // [2]
{
    __shared__ float s0[HD], s1[HD];
    const int d = threadIdx.x;
    const float m = hsum[d] * (1.0f / (float)NNODES);
    s0[d] = m * W_lin[d * 2 + 0];
    s1[d] = m * W_lin[d * 2 + 1];
    __syncthreads();
    if (d == 0) {
        float a = b_lin[0], b = b_lin[1];
        for (int i = 0; i < HD; ++i) { a += s0[i]; b += s1[i]; }
        out[0] = a; out[1] = b;
    }
}

extern "C" void kernel_launch(void* const* d_in, const int* in_sizes, int n_in,
                              void* d_out, int out_size, void* d_ws, size_t ws_size,
                              hipStream_t stream) {
    const float* x_seq  = (const float*)d_in[0];
    // d_in[1]: edge_index (int64) — unused: ChebConv K=1 has no neighbor aggregation
    const float* W_x    = (const float*)d_in[2];
    const float* Theta  = (const float*)d_in[3];
    const float* b_conv = (const float*)d_in[4];
    const float* b_gate = (const float*)d_in[5];
    const float* w_ci   = (const float*)d_in[6];
    const float* w_cf   = (const float*)d_in[7];
    const float* w_co   = (const float*)d_in[8];
    const float* W_lin  = (const float*)d_in[9];
    const float* b_lin  = (const float*)d_in[10];

    float* hsum = (float*)d_ws;  // 64 floats of scratch
    hipMemsetAsync(hsum, 0, HD * sizeof(float), stream);

    gclstm_kernel<<<NNODES / TILE_N, THREADS, 0, stream>>>(
        x_seq, W_x, Theta, b_conv, b_gate, w_ci, w_cf, w_co, hsum);
    finalize_kernel<<<1, HD, 0, stream>>>(hsum, W_lin, b_lin, (float*)d_out);
}
